// AttentiveLayer_43593918054759
// MI455X (gfx1250) — matrix operand order repaired and checked by reading, not compiled, not run on hardware
//
#include <hip/hip_runtime.h>
#include <hip/hip_bf16.h>

typedef __attribute__((ext_vector_type(16))) _Float16 v16h;
typedef __attribute__((ext_vector_type(8)))  _Float16 v8h;
typedef __attribute__((ext_vector_type(8)))  float    v8f;

constexpr int DIM    = 512;  // feature dim (compile-time for tiling)
constexpr int BM     = 64;   // rows per workgroup
constexpr int BD     = 256;  // dims per workgroup (grid.y = DIM/BD)
constexpr int BK     = 32;   // j-chunk (K of one WMMA)
constexpr int NT     = 8;    // 128/16 N-tiles per wave
constexpr int XPITCH = 40;   // halfs per xT/pT row (80B pitch -> conflict-free b128)
constexpr int APITCH = 36;   // ints per adjT row (144B pitch)

// ---------------- kernel 1: wh1/wh2 = x . a[:D], x . a[D:] -------------------
__global__ void prep_wh_kernel(const float* __restrict__ x,
                               const float* __restrict__ a,
                               float* __restrict__ wh1,
                               float* __restrict__ wh2,
                               int n, int dim) {
  int wave = threadIdx.x >> 5, lane = threadIdx.x & 31;
  int row = blockIdx.x * 8 + wave;
  if (row >= n) return;
  const float* xr = x + (size_t)row * dim;
  float s1 = 0.f, s2 = 0.f;
  for (int t = lane; t < dim; t += 32) {
    float xv = xr[t];
    s1 = fmaf(xv, a[t], s1);
    s2 = fmaf(xv, a[dim + t], s2);
  }
#pragma unroll
  for (int off = 16; off >= 1; off >>= 1) {
    s1 += __shfl_xor(s1, off, 32);
    s2 += __shfl_xor(s2, off, 32);
  }
  if (lane == 0) { wh1[row] = s1; wh2[row] = s2; }
}

// ---------------- kernel 2: xhT[d][i] = (f16) x[i][d] ------------------------
__global__ __launch_bounds__(256) void transpose_kernel(const float* __restrict__ x,
                                                        _Float16* __restrict__ xhT,
                                                        int n, int dim) {
  __shared__ _Float16 t[32][33];
  int tx = threadIdx.x & 31, ty = threadIdx.x >> 5;  // 32 x 8
  int c0 = blockIdx.x * 32;  // dim tile
  int r0 = blockIdx.y * 32;  // row tile
#pragma unroll
  for (int k = 0; k < 4; ++k) {
    int r = ty + k * 8;
    t[r][tx] = (_Float16)x[(size_t)(r0 + r) * dim + c0 + tx];
  }
  __syncthreads();
#pragma unroll
  for (int k = 0; k < 4; ++k) {
    int r = ty + k * 8;
    xhT[(size_t)(c0 + r) * n + r0 + tx] = t[tx][r];
  }
}

// lds offset of a __shared__ object = low 32 bits of its generic address
__device__ __forceinline__ unsigned ldsOff(const void* p) {
  return (unsigned)(unsigned long long)p;
}
__device__ __forceinline__ void asyncCopy16(unsigned ldsDst, const void* gsrc) {
  asm volatile("global_load_async_to_lds_b128 %0, %1, off"
               :: "v"(ldsDst), "v"((unsigned long long)gsrc)
               : "memory");
}

// ---------------- kernel 3: fused masked-softmax attention @ x ---------------
// grid = (n/BM, DIM/BD), block = 256 threads (8 waves).
// Per j-chunk: async-DMA adj + x^T slab into double-buffered LDS
// (global_load_async_to_lds_b128, ASYNCcnt) while the previous chunk computes;
// all waves jointly compute P = exp(leaky(wh1_i + wh2_j)) (masked
// entries set to exact 0) into LDS in WMMA-A layout; each wave then issues 8x
// v_wmma_f32_16x16x32_f16. Row sums divide at the end (shift-free softmax).
__global__ __launch_bounds__(256) void gat_fused_kernel(
    const int* __restrict__ adj, const float* __restrict__ wh1,
    const float* __restrict__ wh2, const _Float16* __restrict__ xhT,
    float* __restrict__ out, int n) {
  __shared__ __align__(16) _Float16 xT[2][BD * XPITCH];    // 2 x 20 KB
  __shared__ __align__(16) int      adjT[2][BM * APITCH];  // 2 x  9 KB
  __shared__ __align__(16) _Float16 pT[BM * XPITCH];       //      5 KB
  __shared__ float lrow[BM];

  const int tid  = threadIdx.x;
  const int wave = tid >> 5, lane = tid & 31;
  const int rt = wave & 3;   // consumer row tile (16 rows each)
  const int dq = wave >> 2;  // consumer dim quarter (128 cols each)
  const int lh = lane >> 4, lr = lane & 15;
  const int rowBase = blockIdx.x * BM;
  const int dimBase = blockIdx.y * BD;

  // producer assignment: lane owns (row = wave*8 + lane/4, k in [pkb, pkb+8))
  const int prow = wave * 8 + (lane >> 2);  // local row 0..63
  const int pkb  = (lane & 3) * 8;          // k offset 0,8,16,24
  const float w1 = wh1[rowBase + prow];
  float lacc = 0.f;

  // async stage of one chunk into buffer `buf`: 6 x b128 per thread
  auto stage = [&](int buf, int j0) {
#pragma unroll
    for (int it = 0; it < 4; ++it) {           // x^T: 256 dims x 32 halfs
      int idx = tid + it * 256;                // 0..1023
      int r = idx >> 2, s = idx & 3;
      asyncCopy16(ldsOff(&xT[buf][r * XPITCH + s * 8]),
                  xhT + (size_t)(dimBase + r) * n + j0 + s * 8);
    }
#pragma unroll
    for (int it = 0; it < 2; ++it) {           // adj: 64 rows x 32 ints
      int idx = tid + it * 256;                // 0..511
      int r = idx >> 3, s = idx & 7;
      asyncCopy16(ldsOff(&adjT[buf][r * APITCH + s * 4]),
                  adj + (size_t)(rowBase + r) * n + j0 + s * 4);
    }
  };

  v8f acc[NT] = {};

  const int nChunks = n / BK;
  stage(0, 0);  // prologue

  for (int c = 0; c < nChunks; ++c) {
    const int buf = c & 1;
    const int j0 = c * BK;

    // my async writes into `buf` must have landed before I publish them
    asm volatile("s_wait_asynccnt 0x0" ::: "memory");
    __syncthreads();

    // kick off the next chunk's DMA; it overlaps the P + WMMA phases below
    if (c + 1 < nChunks) stage(buf ^ 1, j0 + BK);

    // ---- P phase: each lane computes 8 elements, once per block ----
    {
      float4 wa = *(const float4*)(wh2 + j0 + pkb);
      float4 wb = *(const float4*)(wh2 + j0 + pkb + 4);
      int4 ma = *(const int4*)&adjT[buf][prow * APITCH + pkb];
      int4 mb = *(const int4*)&adjT[buf][prow * APITCH + pkb + 4];
      float sv[8] = {wa.x, wa.y, wa.z, wa.w, wb.x, wb.y, wb.z, wb.w};
      int   mv[8] = {ma.x, ma.y, ma.z, ma.w, mb.x, mb.y, mb.z, mb.w};
      union { int4 i4v; _Float16 h[8]; } pk;
      float csum = 0.f;
#pragma unroll
      for (int k = 0; k < 8; ++k) {
        float s = w1 + sv[k];
        float e = fmaxf(s, 0.01f * s);               // leaky_relu (slope<1)
        float p = (mv[k] > 0) ? __expf(e) : 0.f;     // mask -> exact 0
        csum += p;
        pk.h[k] = (_Float16)p;
      }
      lacc += csum;
      *(int4*)&pT[prow * XPITCH + pkb] = pk.i4v;     // A-layout friendly
    }
    __syncthreads();

    // ---- WMMA phase ----
    union { v16h v; v8h h[2]; } A, B;
    {
      // A 16x32 f16 layout: lane(row=lr): K {8lh..8lh+7} and {16+8lh..+7}
      const _Float16* ap = &pT[(rt * 16 + lr) * XPITCH];
      A.h[0] = *(const v8h*)(ap + 8 * lh);
      A.h[1] = *(const v8h*)(ap + 16 + 8 * lh);
    }
#pragma unroll
    for (int nt = 0; nt < NT; ++nt) {
      // B 32x16 f16 layout: lane(col=lr): K {8lh..8lh+7} and {16+8lh..+7}
      const _Float16* bp = &xT[buf][(dq * 128 + nt * 16 + lr) * XPITCH];
      B.h[0] = *(const v8h*)(bp + 8 * lh);
      B.h[1] = *(const v8h*)(bp + 16 + 8 * lh);
      acc[nt] = __builtin_amdgcn_wmma_f32_16x16x32_f16(
          false, A.v, false, B.v, (short)0, acc[nt], false, false);
    }
  }

  // ---- row sums: combine the 4 k-quarter lanes of each producer row ----
  lacc += __shfl_xor(lacc, 1, 32);
  lacc += __shfl_xor(lacc, 2, 32);
  if ((lane & 3) == 0) lrow[prow] = lacc;
  __syncthreads();

  // ---- normalize + store (C layout: vgpr v, lane-half -> row v + 8*lh) ----
#pragma unroll
  for (int v = 0; v < 8; ++v) {
    float lv = lrow[rt * 16 + v + 8 * lh];
    float rin = (lv > 0.f) ? (1.0f / lv) : 0.f;
    int orow = rowBase + rt * 16 + v + 8 * lh;
    float* op = out + (size_t)orow * DIM + dimBase + dq * 128 + lr;
#pragma unroll
    for (int nt = 0; nt < NT; ++nt) op[nt * 16] = acc[nt][v] * rin;
  }
}

// ---------------------------------------------------------------------------
extern "C" void kernel_launch(void* const* d_in, const int* in_sizes, int n_in,
                              void* d_out, int out_size, void* d_ws, size_t ws_size,
                              hipStream_t stream) {
  const float* x   = (const float*)d_in[0];
  const int*   adj = (const int*)d_in[1];
  const float* a   = (const float*)d_in[2];
  const int dim = in_sizes[2] / 2;    // 512
  const int n   = in_sizes[0] / dim;  // 8192

  // workspace: wh1[n] f32 | wh2[n] f32 | xhT[dim][n] f16  (~8.06 MB)
  float* wh1 = (float*)d_ws;
  float* wh2 = wh1 + n;
  _Float16* xhT = (_Float16*)(wh2 + n);

  prep_wh_kernel<<<n / 8, 256, 0, stream>>>(x, a, wh1, wh2, n, dim);

  dim3 tg(dim / 32, n / 32);
  transpose_kernel<<<tg, 256, 0, stream>>>(x, xhT, n, dim);

  dim3 gg(n / BM, DIM / BD);
  gat_fused_kernel<<<gg, 256, 0, stream>>>(adj, wh1, wh2, xhT, (float*)d_out, n);
}
